// SSD_66563403153551
// MI455X (gfx1250) — compile-verified, matrix operand order checked by reading, not run on hardware
//
#include <hip/hip_runtime.h>
#include <hip/hip_bf16.h>
#include <math.h>

typedef __attribute__((ext_vector_type(16))) _Float16 v16h;
typedef __attribute__((ext_vector_type(8)))  float    v8f;

#define NPRIOR 8732
#define TOPK   200
#define BATCH  64

// gfx1250 async global->LDS path, guarded so the file compiles on any toolchain.
// Probe round 3 showed the builtin takes (int4 AS1*, int4 AS3*, imm, imm).
#if defined(__has_builtin)
#if __has_builtin(__builtin_amdgcn_global_load_async_to_lds_b128) && \
    __has_builtin(__builtin_amdgcn_s_wait_asynccnt)
#define USE_ASYNC_LDS 1
typedef int v4i_vec __attribute__((vector_size(16)));
typedef __attribute__((address_space(1))) v4i_vec* gas_v4i_p;
typedef __attribute__((address_space(3))) v4i_vec* las_v4i_p;
#endif
#endif

// ---------------------------------------------------------------------------
// Generic conv as implicit GEMM on the f16 WMMA pipe, fp32 accumulate.
// M = B*Ho*Wo (rows), N = Cout (cols), K = Cin*Ksz*Ksz (padded to mult of 32).
// Block = 128 threads (4 waves); block computes a 16(M) x 64(N) tile, wave w
// owns N sub-tile w. A (16x32, f16) and B (64x32, f32) K-chunks are staged in
// LDS in WMMA fragment layout with double buffering: chunk i+1 is fetched
// (async-to-LDS where available) while chunk i runs on the WMMA pipe.
// ---------------------------------------------------------------------------
__global__ __launch_bounds__(128)
void conv_wmma_kernel(const float* __restrict__ in, const float* __restrict__ wgt,
                      const float* __restrict__ bias, float* __restrict__ out,
                      int Cin, int Hin, int Win, int Cout, int Ho, int Wo,
                      int Ksz, int stride, int pad, int relu, int M, int HWo)
{
    __shared__ __align__(32) _Float16 ldsA[2][512];      // 16 rows x 32 k (fragment layout)
    __shared__ __align__(16) float    ldsB[2][4][512];   // 4 N-tiles x (32 lanes x 16 f32)
    __shared__ int rowB[16], rowOH[16], rowOW[16];       // per-M-row descriptors

    const int lane  = threadIdx.x;                       // 0..31
    const int wv    = threadIdx.y;                       // 0..3 (N sub-tile)
    const int t     = wv * 32 + lane;                    // 0..127
    const int tileM = blockIdx.x;
    const int nbase = blockIdx.y * 64;

    // decompose the 16 M rows once per block
    if (t < 16) {
        int m = tileM * 16 + t;
        if (m < M) {
            int b  = m / HWo;
            int hw = m - b * HWo;
            rowB[t]  = b;
            rowOH[t] = hw / Wo;
            rowOW[t] = hw - (hw / Wo) * Wo;
        } else {
            rowB[t] = -1; rowOH[t] = 0; rowOW[t] = 0;
        }
    }

    const int KK   = Ksz * Ksz;
    const int Ktot = Cin * KK;

    // this thread's weight row for staging B: n = t>>1 in [0,64), half-chunk (t&1)*16
    const int  nloc = t >> 1;
    const int  kk0  = (t & 1) * 16;
    const int  ng   = nbase + nloc;
    const float* wp = (ng < Cout) ? (wgt + (size_t)ng * Ktot) : (const float*)0;

    // ---- staging helpers (fragment layout: lane = row/col + 16*((k>>3)&1),
    //      slot = (k&7) + 8*(k>>4); each aligned 8-k group is 8 contiguous slots) ----
    auto stageA = [&](int bufI, int k0) {
#pragma unroll
        for (int j = 0; j < 4; ++j) {
            int ea  = t + 128 * j;
            int row = ea & 15;
            int kk  = ea >> 4;
            int k   = k0 + kk;
            float fa = 0.0f;
            int b = rowB[row];
            if (b >= 0 && k < Ktot) {
                int c  = k / KK;
                int r  = k - c * KK;
                int ky = r / Ksz;
                int kx = r - ky * Ksz;
                int ih = rowOH[row] * stride - pad + ky;
                int iw = rowOW[row] * stride - pad + kx;
                if ((unsigned)ih < (unsigned)Hin && (unsigned)iw < (unsigned)Win)
                    fa = in[(((size_t)b * Cin + c) * Hin + ih) * Win + iw];
            }
            int lA = row + 16 * ((kk >> 3) & 1);
            int pA = (kk & 7) + 8 * (kk >> 4);
            ldsA[bufI][lA * 16 + pA] = (_Float16)fa;
        }
    };
    auto stageB = [&](int bufI, int k0) {
#ifdef USE_ASYNC_LDS
        if (wp && (k0 + 32) <= Ktot) {
            // 16 contiguous f32 from the weight row -> two 8-slot contiguous LDS
            // runs, each = 2x async b128 (16B src -> 16B LDS), ASYNCcnt-tracked
#pragma unroll
            for (int g = 0; g < 2; ++g) {
                int kkg  = kk0 + 8 * g;
                int lB   = (nloc & 15) + 16 * ((kkg >> 3) & 1);
                int slot = 8 * (kkg >> 4);
                float* dst  = &ldsB[bufI][nloc >> 4][lB * 16 + slot];
                float* srcp = const_cast<float*>(wp + k0 + kkg);
                __builtin_amdgcn_global_load_async_to_lds_b128(
                    (gas_v4i_p)srcp, (las_v4i_p)dst, 0, 0);
                __builtin_amdgcn_global_load_async_to_lds_b128(
                    (gas_v4i_p)(srcp + 4), (las_v4i_p)(dst + 4), 0, 0);
            }
            return;
        }
#endif
        // scalar fallback: tail chunk / padded-N rows (zero fill)
#pragma unroll
        for (int e = 0; e < 16; ++e) {
            int kk = kk0 + e;
            int k  = k0 + kk;
            float fb = (wp && k < Ktot) ? wp[k] : 0.0f;
            int lB = (nloc & 15) + 16 * ((kk >> 3) & 1);
            int pB = (kk & 7) + 8 * (kk >> 4);
            ldsB[bufI][nloc >> 4][lB * 16 + pB] = fb;
        }
    };

    v8f acc = {};
    const int nchunks = (Ktot + 31) >> 5;

    __syncthreads();               // row descriptors ready
    stageA(0, 0);
    stageB(0, 0);

    for (int c = 0; c < nchunks; ++c) {
        const int k0 = c << 5;
#ifdef USE_ASYNC_LDS
        __builtin_amdgcn_s_wait_asynccnt(0);   // my async stores to LDS landed
#endif
        __syncthreads();                       // everyone's stage of chunk c visible

        if (c + 1 < nchunks) {                 // issue chunk c+1 while c computes
            stageA((c + 1) & 1, k0 + 32);
            stageB((c + 1) & 1, k0 + 32);
            if (wp && (k0 + 64) < Ktot)        // global_prefetch_b8 for chunk c+2
                __builtin_prefetch(wp + k0 + 64 + kk0, 0, 1);
        }

        // one aligned 32B LDS read for A; 64B (4x b128) + cvt for B
        v16h av = ((const v16h*)ldsA[c & 1])[lane];
        v16h bv;
        const float* bp = &ldsB[c & 1][wv][lane * 16];
#pragma unroll
        for (int e = 0; e < 16; ++e) bv[e] = (_Float16)bp[e];

        acc = __builtin_amdgcn_wmma_f32_16x16x32_f16(
            /*neg_a=*/false, av, /*neg_b=*/false, bv,
            /*c_mod=*/(short)0, acc, /*reuse_a=*/false, /*reuse_b=*/false);
    }

    // D layout: lane%16 = N col; VGPR r -> M = r (lanes 0-15) or r+8 (lanes 16-31)
    const int half = lane >> 4;
    const int idx  = lane & 15;
    const int ncol = nbase + wv * 16 + idx;
    if (ncol < Cout) {
        const float bsv = bias[ncol];
#pragma unroll
        for (int r2 = 0; r2 < 8; ++r2) {
            const int rrow = half * 8 + r2;
            const int b2 = rowB[rrow];
            if (b2 >= 0) {
                const int hw2 = rowOH[rrow] * Wo + rowOW[rrow];
                float v2 = acc[r2] + bsv;
                if (relu) v2 = fmaxf(v2, 0.0f);
                out[((size_t)b2 * Cout + ncol) * HWo + hw2] = v2;
            }
        }
    }
}

// ---------------------------------------------------------------------------
// Maxpool kxk, stride k, VALID
// ---------------------------------------------------------------------------
__global__ void maxpool_kernel(const float* __restrict__ in, float* __restrict__ out,
                               int C, int Hin, int Win, int k, int Ho, int Wo, long long total)
{
    long long i = (long long)blockIdx.x * blockDim.x + threadIdx.x;
    if (i >= total) return;
    int ow = (int)(i % Wo);
    int oh = (int)((i / Wo) % Ho);
    long long bc = i / ((long long)Wo * Ho);   // b*C + c
    const float* src = in + bc * Hin * Win;
    float m = -INFINITY;
    for (int dy = 0; dy < k; ++dy)
        for (int dx = 0; dx < k; ++dx)
            m = fmaxf(m, src[(oh * k + dy) * Win + (ow * k + dx)]);
    out[i] = m;
}

// ---------------------------------------------------------------------------
// L2Norm over channels (C=32) * learned scale
// ---------------------------------------------------------------------------
__global__ void l2norm_kernel(const float* __restrict__ in, float* __restrict__ out,
                              const float* __restrict__ scale, int C, int HW, long long total)
{
    long long i = (long long)blockIdx.x * blockDim.x + threadIdx.x;  // b*HW + hw
    if (i >= total) return;
    long long b  = i / HW;
    long long hw = i - b * HW;
    const float* src = in + (b * C) * HW + hw;
    float s = 0.0f;
    for (int c = 0; c < C; ++c) { float v = src[(long long)c * HW]; s += v * v; }
    float inv = 1.0f / (sqrtf(s) + 1e-10f);
    float* dst = out + (b * C) * HW + hw;
    for (int c = 0; c < C; ++c)
        dst[(long long)c * HW] = src[(long long)c * HW] * inv * scale[c];
}

// ---------------------------------------------------------------------------
// NCHW -> per-batch (h,w,c) flat, placed at `off` inside a rowTotal-sized row
// (implements transpose(0,2,3,1) + concat)
// ---------------------------------------------------------------------------
__global__ void gather_head_kernel(const float* __restrict__ src, float* __restrict__ dst,
                                   int C, int H, int W, int rowTotal, int off, long long total)
{
    long long i = (long long)blockIdx.x * blockDim.x + threadIdx.x;
    if (i >= total) return;
    int w = (int)(i % W);
    int h = (int)((i / W) % H);
    int c = (int)((i / ((long long)W * H)) % C);
    long long b = i / ((long long)W * H * C);
    dst[b * rowTotal + off + ((long long)h * W + w) * C + c] = src[i];
}

// ---------------------------------------------------------------------------
// Per-image detection: softmax -> threshold -> top-200 -> decode -> NMS -> pack
// ---------------------------------------------------------------------------
__global__ __launch_bounds__(256)
void detect_kernel(const float* __restrict__ locb, const float* __restrict__ confb,
                   const float* __restrict__ dbox, float* __restrict__ out)
{
    __shared__ float sc[NPRIOR];
    __shared__ float rv[256];
    __shared__ int   ri[256];
    __shared__ float cand_s[TOPK];
    __shared__ int   cand_i[TOPK];
    __shared__ float bx[TOPK][4];
    __shared__ float area[TOPK];
    __shared__ int   validf[TOPK];
    __shared__ int   supp[TOPK];

    const int b = blockIdx.x;
    const int t = threadIdx.x;
    const int nt = blockDim.x;

    // class-1 softmax prob == sigmoid(c1 - c0); apply CONF_THRESH masking
    for (int i = t; i < NPRIOR; i += nt) {
        float c0 = confb[((long long)b * NPRIOR + i) * 2 + 0];
        float c1 = confb[((long long)b * NPRIOR + i) * 2 + 1];
        float p  = 1.0f / (1.0f + expf(c0 - c1));
        sc[i] = (p > 0.01f) ? p : -1.0f;
    }
    __syncthreads();

    // top-K by iterative parallel argmax (descending, low-index tie-break)
    for (int k = 0; k < TOPK; ++k) {
        float bvv = -1.0e30f; int bii = NPRIOR;
        for (int i = t; i < NPRIOR; i += nt) {
            float v = sc[i];
            if (v > bvv || (v == bvv && i < bii)) { bvv = v; bii = i; }
        }
        rv[t] = bvv; ri[t] = bii;
        __syncthreads();
        for (int s = 128; s > 0; s >>= 1) {
            if (t < s) {
                float v2 = rv[t + s]; int i2 = ri[t + s];
                if (v2 > rv[t] || (v2 == rv[t] && i2 < ri[t])) { rv[t] = v2; ri[t] = i2; }
            }
            __syncthreads();
        }
        if (t == 0) {
            cand_s[k] = rv[0];
            cand_i[k] = ri[0];
            sc[ri[0]] = -2.0f;   // remove from further rounds
        }
        __syncthreads();
    }

    // decode candidate boxes
    if (t < TOPK) {
        int i = cand_i[t];
        float l0 = locb[((long long)b * NPRIOR + i) * 4 + 0];
        float l1 = locb[((long long)b * NPRIOR + i) * 4 + 1];
        float l2 = locb[((long long)b * NPRIOR + i) * 4 + 2];
        float l3 = locb[((long long)b * NPRIOR + i) * 4 + 3];
        float d0 = dbox[i * 4 + 0], d1 = dbox[i * 4 + 1];
        float d2 = dbox[i * 4 + 2], d3 = dbox[i * 4 + 3];
        float cx = d0 + l0 * 0.1f * d2;
        float cy = d1 + l1 * 0.1f * d3;
        float w  = d2 * expf(l2 * 0.2f);
        float h  = d3 * expf(l3 * 0.2f);
        float x0 = cx - w * 0.5f, y0 = cy - h * 0.5f;
        bx[t][0] = x0; bx[t][1] = y0; bx[t][2] = x0 + w; bx[t][3] = y0 + h;
        area[t]  = w * h;
        validf[t] = (cand_s[t] > 0.01f) ? 1 : 0;
        supp[t]   = 0;
    }
    __syncthreads();

    // sequential greedy NMS, exactly mirroring the reference fori_loop
    for (int k = 0; k < TOPK; ++k) {
        bool keep_k = (!supp[k]) && validf[k];
        if (keep_k && t < TOPK && t > k) {
            float lx = fminf(fmaxf(bx[t][0], bx[k][0]), bx[k][2]);
            float hx = fminf(fmaxf(bx[t][2], bx[k][0]), bx[k][2]);
            float ly = fminf(fmaxf(bx[t][1], bx[k][1]), bx[k][3]);
            float hy = fminf(fmaxf(bx[t][3], bx[k][1]), bx[k][3]);
            float inter = (hx - lx) * (hy - ly);
            float iou = inter / (area[t] - inter + area[k]);
            if (iou > 0.45f) supp[t] = 1;
        }
        __syncthreads();
    }

    // compact kept rows into out[b][1][pos][0:5]
    if (t == 0) {
        int pos = 0;
        for (int k = 0; k < TOPK; ++k) {
            if (!supp[k] && validf[k]) {
                float* o = out + (((long long)b * 2 + 1) * TOPK + pos) * 5;
                o[0] = cand_s[k];
                o[1] = bx[k][0]; o[2] = bx[k][1]; o[3] = bx[k][2]; o[4] = bx[k][3];
                ++pos;
            }
        }
    }
}

// ---------------------------------------------------------------------------
// Host-side launch plumbing
// ---------------------------------------------------------------------------
static inline void launch_conv(const float* in, const float* w, const float* bias, float* out,
                               int Bn, int Cin, int Hin, int Win, int Cout,
                               int Ksz, int stride, int pad, int relu, hipStream_t s)
{
    int Ho = (Hin + 2 * pad - Ksz) / stride + 1;
    int Wo = (Win + 2 * pad - Ksz) / stride + 1;
    int M = Bn * Ho * Wo;
    dim3 grid((M + 15) / 16, (Cout + 63) / 64);
    dim3 blk(32, 4);
    conv_wmma_kernel<<<grid, blk, 0, s>>>(in, w, bias, out, Cin, Hin, Win, Cout,
                                          Ho, Wo, Ksz, stride, pad, relu, M, Ho * Wo);
}

static inline void launch_mp(const float* in, float* out, int Bn, int C, int Hin, int k, hipStream_t s)
{
    int Ho = Hin / k;
    long long total = (long long)Bn * C * Ho * Ho;
    maxpool_kernel<<<(unsigned)((total + 255) / 256), 256, 0, s>>>(in, out, C, Hin, Hin, k, Ho, Ho, total);
}

extern "C" void kernel_launch(void* const* d_in, const int* in_sizes, int n_in,
                              void* d_out, int out_size, void* d_ws, size_t ws_size,
                              hipStream_t stream)
{
    (void)in_sizes; (void)n_in; (void)ws_size;
    // --- unpack inputs in setup_inputs() dict order ---
    int p = 0;
    const float* x = (const float*)d_in[p++];
    const float *vw[7], *vb[7];
    for (int i = 0; i < 7; ++i) { vw[i] = (const float*)d_in[p++]; vb[i] = (const float*)d_in[p++]; }
    const float *ew[8], *eb[8];
    for (int i = 0; i < 8; ++i) { ew[i] = (const float*)d_in[p++]; eb[i] = (const float*)d_in[p++]; }
    const float *lw[6], *lb[6];
    for (int i = 0; i < 6; ++i) { lw[i] = (const float*)d_in[p++]; lb[i] = (const float*)d_in[p++]; }
    const float *cw[6], *cb[6];
    for (int i = 0; i < 6; ++i) { cw[i] = (const float*)d_in[p++]; cb[i] = (const float*)d_in[p++]; }
    const float* l2n  = (const float*)d_in[p++];
    const float* dbox = (const float*)d_in[p++];

    // --- workspace layout (floats): two ping-pong arenas + persistent buffers ---
    float* wsf = (float*)d_ws;
    size_t off = 0;
    float* A  = wsf + off; off += 46080000ULL;   // max conv out: 64*8*300*300
    float* Bb = wsf + off; off += 10240000ULL;   // max alt out: 64*16*100*100
    float* S0 = wsf + off; off += (size_t)BATCH * 32 * 38 * 38;
    float* S1 = wsf + off; off += (size_t)BATCH * 32 * 19 * 19;
    float* S2 = wsf + off; off += (size_t)BATCH * 64 * 10 * 10;
    float* S3 = wsf + off; off += (size_t)BATCH * 32 * 5 * 5;
    float* S4 = wsf + off; off += (size_t)BATCH * 32 * 3 * 3;
    float* S5 = wsf + off; off += (size_t)BATCH * 32 * 1 * 1;
    float* LOCB  = wsf + off; off += (size_t)BATCH * NPRIOR * 4;
    float* CONFB = wsf + off; off += (size_t)BATCH * NPRIOR * 2;

    const int B = BATCH;

    // --- VGG backbone ---
    launch_conv(x,  vw[0], vb[0], A,  B, 2, 300, 300,  8, 3, 1, 1, 1, stream);  // 8x300x300
    launch_mp(A, Bb, B, 8, 300, 3, stream);                                     // 8x100x100
    launch_conv(Bb, vw[1], vb[1], A,  B, 8, 100, 100, 16, 3, 1, 1, 1, stream);  // 16x100x100
    launch_mp(A, Bb, B, 16, 100, 2, stream);                                    // 16x50x50
    launch_conv(Bb, vw[2], vb[2], A,  B, 16, 50, 50, 32, 5, 1, 0, 1, stream);   // 32x46x46
    launch_conv(A,  vw[3], vb[3], Bb, B, 32, 46, 46, 32, 5, 1, 0, 1, stream);   // 32x42x42
    launch_conv(Bb, vw[4], vb[4], A,  B, 32, 42, 42, 32, 5, 1, 0, 1, stream);   // 32x38x38

    {   // L2Norm source 0
        long long total = (long long)B * 38 * 38;
        l2norm_kernel<<<(unsigned)((total + 255) / 256), 256, 0, stream>>>(A, S0, l2n, 32, 38 * 38, total);
    }
    launch_mp(A, Bb, B, 32, 38, 2, stream);                                     // 32x19x19
    launch_conv(Bb, vw[5], vb[5], A,  B, 32, 19, 19, 32, 3, 1, 1, 1, stream);   // 32x19x19
    launch_conv(A,  vw[6], vb[6], S1, B, 32, 19, 19, 32, 3, 1, 1, 1, stream);   // source 1

    // --- extras ---
    launch_conv(S1, ew[0], eb[0], A,  B,  32, 19, 19, 256, 1, 1, 0, 1, stream); // 256x19x19
    launch_conv(A,  ew[1], eb[1], S2, B, 256, 19, 19,  64, 3, 2, 1, 1, stream); // source 2: 64x10x10
    launch_conv(S2, ew[2], eb[2], A,  B,  64, 10, 10,  64, 1, 1, 0, 1, stream);
    launch_conv(A,  ew[3], eb[3], S3, B,  64, 10, 10,  32, 3, 2, 1, 1, stream); // source 3: 32x5x5
    launch_conv(S3, ew[4], eb[4], A,  B,  32,  5,  5,  64, 1, 1, 0, 1, stream);
    launch_conv(A,  ew[5], eb[5], S4, B,  64,  5,  5,  32, 3, 1, 0, 1, stream); // source 4: 32x3x3
    launch_conv(S4, ew[6], eb[6], A,  B,  32,  3,  3,  32, 1, 1, 0, 1, stream);
    launch_conv(A,  ew[7], eb[7], S5, B,  32,  3,  3,  32, 3, 1, 0, 1, stream); // source 5: 32x1x1

    // --- detection heads + concat (transpose to NHWC, flatten, concat) ---
    const float* srcs[6] = { S0, S1, S2, S3, S4, S5 };
    const int srcC[6]  = { 32, 32, 64, 32, 32, 32 };
    const int srcH[6]  = { 38, 19, 10,  5,  3,  1 };
    const int headK[6] = {  3,  3,  3,  3,  3,  1 };
    const int headP[6] = {  1,  1,  1,  1,  1,  0 };
    const int asp[6]   = {  4,  6,  6,  6,  4,  4 };
    const int locRow = NPRIOR * 4, confRow = NPRIOR * 2;
    int locOff = 0, confOff = 0;
    for (int i = 0; i < 6; ++i) {
        int H = srcH[i];
        int lc = asp[i] * 4, cc = asp[i] * 2;
        // loc head
        launch_conv(srcs[i], lw[i], lb[i], Bb, B, srcC[i], H, H, lc, headK[i], 1, headP[i], 0, stream);
        long long tl = (long long)B * lc * H * H;
        gather_head_kernel<<<(unsigned)((tl + 255) / 256), 256, 0, stream>>>(Bb, LOCB, lc, H, H, locRow, locOff, tl);
        // conf head
        launch_conv(srcs[i], cw[i], cb[i], Bb, B, srcC[i], H, H, cc, headK[i], 1, headP[i], 0, stream);
        long long tc = (long long)B * cc * H * H;
        gather_head_kernel<<<(unsigned)((tc + 255) / 256), 256, 0, stream>>>(Bb, CONFB, cc, H, H, confRow, confOff, tc);
        locOff  += H * H * lc;
        confOff += H * H * cc;
    }

    // --- detect: softmax + decode + NMS + pack; class-0 plane is zeros ---
    (void)hipMemsetAsync(d_out, 0, (size_t)out_size * sizeof(float), stream);
    detect_kernel<<<B, 256, 0, stream>>>(LOCB, CONFB, dbox, (float*)d_out);
}
